// PILONTransformer_25666724560979
// MI455X (gfx1250) — matrix operand order repaired and checked by reading, not compiled, not run on hardware
//
#include <hip/hip_runtime.h>
#include <hip/hip_bf16.h>
#include <stdint.h>

// ---------------------------------------------------------------------------
// Types / helpers for CDNA5 WMMA (wave32, 16x16x32 bf16 -> f32)
// ---------------------------------------------------------------------------
typedef __bf16 bf16;
typedef bf16  v16bf __attribute__((ext_vector_type(16)));
typedef bf16  v8bf  __attribute__((ext_vector_type(8)));
typedef bf16  v4bf  __attribute__((ext_vector_type(4)));
typedef float v8f   __attribute__((ext_vector_type(8)));

union frag16 { v16bf v; v8bf h[2]; };

__device__ __forceinline__ bf16 f2bf(float f) { return (bf16)f; }

__device__ __forceinline__ v8f wmma_bf16(v16bf a, v16bf b, v8f c) {
  return __builtin_amdgcn_wmma_f32_16x16x32_bf16(
      false, a, false, b, (short)0, c, false, false);
}

__device__ __forceinline__ v8bf ld16B(const bf16* p) {
  return *(const v8bf*)__builtin_assume_aligned(p, 16);
}
__device__ __forceinline__ void st16B(bf16* p, v8bf v) {
  *(v8bf*)__builtin_assume_aligned(p, 16) = v;
}
__device__ __forceinline__ void st8B(bf16* p, v4bf v) {
  *(v4bf*)__builtin_assume_aligned(p, 8) = v;
}

// A fragment: 16x32 bf16 tile, row-major (ld elements/row). Two ds_load_b128.
__device__ __forceinline__ v16bf load_a_frag(const bf16* s, int ld) {
  int lane = threadIdx.x & 31;
  int m = lane & 15;
  int lo = (lane < 16) ? 0 : 16;
  const bf16* row = s + m * ld;
  frag16 f;
  f.h[0] = ld16B(row + lo);
  f.h[1] = ld16B(row + lo + 8);
  return f.v;
}

// B fragment from fragment-major staged tile (lane's 16 bf16 contiguous).
#define LSTR 24
__device__ __forceinline__ v16bf load_bf_frag(const bf16* s) {
  int lane = threadIdx.x & 31;
  const bf16* p = s + lane * LSTR;
  frag16 f;
  f.h[0] = ld16B(p);
  f.h[1] = ld16B(p + 8);
  return f.v;
}

// B = S^T fragment, S row-major [n][k] (K tile [key][dh]). Two ds_load_b128.
__device__ __forceinline__ v16bf load_bT_frag(const bf16* s, int ld) {
  int lane = threadIdx.x & 31;
  int n = lane & 15;
  int kb = (lane < 16) ? 0 : 16;
  const bf16* p = s + n * ld + kb;
  frag16 f;
  f.h[0] = ld16B(p);
  f.h[1] = ld16B(p + 8);
  return f.v;
}

// fragment-major lane index for element (k, n) of a 32xBN tile
#define BF_LANE(k, n) ((((k) & 16)) | ((n) & 15))

// ---------------------------------------------------------------------------
// Problem constants
// ---------------------------------------------------------------------------
#define BB 2
#define SS 512
#define DD 1024
#define HH 16
#define DH 64
#define PP 16
#define KK 4
#define FF 4096
#define NTOK (BB * SS)
#define NENT (NTOK * KK)

// ---------------------------------------------------------------------------
// RMSNorm
// ---------------------------------------------------------------------------
__global__ __launch_bounds__(256) void k_rmsnorm(const float* __restrict__ x,
                                                 const float* __restrict__ g,
                                                 float* __restrict__ out) {
  __shared__ float red[8];
  const int t = blockIdx.x;
  const float* row = x + (size_t)t * DD;
  float ss = 0.f;
  for (int i = threadIdx.x; i < DD; i += 256) { float v = row[i]; ss += v * v; }
#pragma unroll
  for (int off = 16; off; off >>= 1) ss += __shfl_xor(ss, off, 32);
  if ((threadIdx.x & 31) == 0) red[threadIdx.x >> 5] = ss;
  __syncthreads();
  if (threadIdx.x < 8) {
    float v = red[threadIdx.x];
#pragma unroll
    for (int off = 4; off; off >>= 1) v += __shfl_xor(v, off, 8);
    if (threadIdx.x == 0) red[0] = v;
  }
  __syncthreads();
  const float inv = rsqrtf(red[0] * (1.0f / DD) + 1e-6f);
  for (int i = threadIdx.x; i < DD; i += 256)
    out[(size_t)t * DD + i] = row[i] * inv * g[i];
}

// ---------------------------------------------------------------------------
// Dense bf16-WMMA GEMM: C[M,N] = A@B (+R). 128x128 tile, BK=32, 8 waves,
// 8 accumulators / wave (2 row-frags x 4 col-frags).
// ---------------------------------------------------------------------------
__global__ __launch_bounds__(256) void k_gemm(const float* __restrict__ A,
                                              const float* __restrict__ Bw,
                                              const float* __restrict__ R,
                                              float* __restrict__ C,
                                              int M, int N, int Kd) {
  __shared__ bf16 sA[128][32];
  __shared__ bf16 sBf[8][32][LSTR];
  const int tid = threadIdx.x;
  const int wave = tid >> 5;
  const int wm = wave & 3;   // 32-row group
  const int wn = wave >> 2;  // 64-col group
  const int bm = blockIdx.y * 128;
  const int bn = blockIdx.x * 128;
  v8f acc[2][4] = {};
  for (int k0 = 0; k0 < Kd; k0 += 32) {
    __syncthreads();
    {  // A: one row x 16 consecutive elements per thread -> 2x 16B ds_store
      int row = tid >> 1, kc = (tid & 1) * 16;
      const float* src = A + (size_t)(bm + row) * Kd + k0 + kc;
      v8bf t0, t1;
#pragma unroll
      for (int j = 0; j < 8; ++j) { t0[j] = f2bf(src[j]); t1[j] = f2bf(src[8 + j]); }
      st16B(&sA[row][kc], t0);
      st16B(&sA[row][kc + 8], t1);
    }
    {  // B: 4k x 4n per thread, fragment-major, 4x ds_store_b64
      int kg = (tid >> 5) * 4;
      int ng = (tid & 31) * 4;
      const float* src = Bw + (size_t)(k0 + kg) * N + bn + ng;
      if (k0 + 32 < Kd) __builtin_prefetch(src + (size_t)32 * N, 0, 3);
      float ld[4][4];
#pragma unroll
      for (int i = 0; i < 4; ++i) {
        float4 v = *(const float4*)(src + (size_t)i * N);
        ld[i][0] = v.x; ld[i][1] = v.y; ld[i][2] = v.z; ld[i][3] = v.w;
      }
#pragma unroll
      for (int j = 0; j < 4; ++j) {
        int n = ng + j;
        v4bf t;
#pragma unroll
        for (int i = 0; i < 4; ++i) t[i] = f2bf(ld[i][j]);
        st8B(&sBf[n >> 4][BF_LANE(kg, n)][kg & 15], t);
      }
    }
    __syncthreads();
    v16bf a0 = load_a_frag(&sA[wm * 32][0], 32);
    v16bf a1 = load_a_frag(&sA[wm * 32 + 16][0], 32);
#pragma unroll
    for (int t = 0; t < 4; ++t) {
      v16bf b = load_bf_frag(&sBf[wn * 4 + t][0][0]);
      acc[0][t] = wmma_bf16(a0, b, acc[0][t]);
      acc[1][t] = wmma_bf16(a1, b, acc[1][t]);
    }
  }
  const int lane = tid & 31;
  const int n = lane & 15;
  const int rbase = (lane < 16) ? 0 : 8;
#pragma unroll
  for (int f = 0; f < 2; ++f) {
#pragma unroll
    for (int t = 0; t < 4; ++t) {
#pragma unroll
      for (int r = 0; r < 8; ++r) {
        int m = bm + wm * 32 + f * 16 + rbase + r;
        size_t i0 = (size_t)m * N + bn + wn * 64 + t * 16 + n;
        float v = acc[f][t][r];
        if (R) v += R[i0];
        C[i0] = v;
      }
    }
  }
}

// ---------------------------------------------------------------------------
// Flash attention (causal), 4 waves x 16 query rows, WMMA QK^T and PV.
// ---------------------------------------------------------------------------
__global__ __launch_bounds__(128) void k_attn(const float* __restrict__ q,
                                              const float* __restrict__ kk,
                                              const float* __restrict__ vv,
                                              float* __restrict__ o) {
  __shared__ bf16 sQ[64][64];
  __shared__ bf16 sK[32][64];
  __shared__ bf16 sVf[4][32][LSTR];
  __shared__ bf16 sP[4][16][32];
  const int qt = blockIdx.x, h = blockIdx.y, b = blockIdx.z;
  const int tid = threadIdx.x, wave = tid >> 5, lane = tid & 31;
  const int n = lane & 15, rbase = (lane < 16) ? 0 : 8;
  const size_t hd = (size_t)h * DH;
  const float scale = 0.125f;

  {  // stage Q: one row x 32 elements per thread
    int r = tid >> 1, c0 = (tid & 1) * 32;
    const float* src = q + ((size_t)(b * SS + qt * 64 + r)) * DD + hd + c0;
    v8bf t;
#pragma unroll
    for (int c = 0; c < 32; c += 8) {
#pragma unroll
      for (int j = 0; j < 8; ++j) t[j] = f2bf(src[c + j]);
      st16B(&sQ[r][c0 + c], t);
    }
  }
  __syncthreads();
  v16bf aq0 = load_a_frag(&sQ[wave * 16][0], 64);
  v16bf aq1 = load_a_frag(&sQ[wave * 16][32], 64);

  v8f accO[4] = {};
  float rmax[8], rsum[8];
#pragma unroll
  for (int r = 0; r < 8; ++r) { rmax[r] = -__builtin_inff(); rsum[r] = 0.f; }

  const int kend = qt * 64 + 64;
  for (int j0 = 0; j0 < kend; j0 += 32) {
    __syncthreads();
    {  // stage K (row-major) and V (fragment-major): 4r x 4c per thread
      int rg = (tid >> 4) * 4;   // key base
      int cg = (tid & 15) * 4;   // dh base
      const float* kp = kk + ((size_t)(b * SS + j0 + rg)) * DD + hd + cg;
      const float* vp = vv + ((size_t)(b * SS + j0 + rg)) * DD + hd + cg;
      float lk[4][4], lv[4][4];
#pragma unroll
      for (int i = 0; i < 4; ++i) {
        float4 a = *(const float4*)(kp + (size_t)i * DD);
        float4 c = *(const float4*)(vp + (size_t)i * DD);
        lk[i][0] = a.x; lk[i][1] = a.y; lk[i][2] = a.z; lk[i][3] = a.w;
        lv[i][0] = c.x; lv[i][1] = c.y; lv[i][2] = c.z; lv[i][3] = c.w;
      }
#pragma unroll
      for (int i = 0; i < 4; ++i) {  // K rows contiguous in dh
        v4bf t;
#pragma unroll
        for (int j = 0; j < 4; ++j) t[j] = f2bf(lk[i][j]);
        st8B(&sK[rg + i][cg], t);
      }
#pragma unroll
      for (int j = 0; j < 4; ++j) {  // V fragment-major (pack over keys)
        int c = cg + j;
        v4bf t;
#pragma unroll
        for (int i = 0; i < 4; ++i) t[i] = f2bf(lv[i][j]);
        st8B(&sVf[c >> 4][BF_LANE(rg, c)][rg & 15], t);
      }
    }
    __syncthreads();
    v8f s0 = {}, s1 = {};
    {
      v16bf b00 = load_bT_frag(&sK[0][0], 64);
      v16bf b01 = load_bT_frag(&sK[0][32], 64);
      v16bf b10 = load_bT_frag(&sK[16][0], 64);
      v16bf b11 = load_bT_frag(&sK[16][32], 64);
      s0 = wmma_bf16(aq0, b00, s0); s0 = wmma_bf16(aq1, b01, s0);
      s1 = wmma_bf16(aq0, b10, s1); s1 = wmma_bf16(aq1, b11, s1);
    }
#pragma unroll
    for (int r = 0; r < 8; ++r) {
      int qg = qt * 64 + wave * 16 + rbase + r;
      int k0i = j0 + n, k1i = j0 + 16 + n;
      float v0 = (k0i <= qg) ? s0[r] * scale : -__builtin_inff();
      float v1 = (k1i <= qg) ? s1[r] * scale : -__builtin_inff();
      float tm = fmaxf(v0, v1);
#pragma unroll
      for (int off = 8; off; off >>= 1) tm = fmaxf(tm, __shfl_xor(tm, off, 32));
      float mnew = fmaxf(rmax[r], tm);
      float p0 = __expf(v0 - mnew), p1 = __expf(v1 - mnew);
      float ps = p0 + p1;
#pragma unroll
      for (int off = 8; off; off >>= 1) ps += __shfl_xor(ps, off, 32);
      float sc = __expf(rmax[r] - mnew);
      rsum[r] = rsum[r] * sc + ps;
      rmax[r] = mnew;
#pragma unroll
      for (int t = 0; t < 4; ++t) accO[t][r] *= sc;
      sP[wave][rbase + r][n] = f2bf(p0);
      sP[wave][rbase + r][16 + n] = f2bf(p1);
    }
    __syncthreads();
    v16bf ap = load_a_frag(&sP[wave][0][0], 32);
#pragma unroll
    for (int t = 0; t < 4; ++t) {
      v16bf bv = load_bf_frag(&sVf[t][0][0]);
      accO[t] = wmma_bf16(ap, bv, accO[t]);
    }
  }
#pragma unroll
  for (int r = 0; r < 8; ++r) {
    int qg = qt * 64 + wave * 16 + rbase + r;
    float inv = 1.0f / rsum[r];
#pragma unroll
    for (int t = 0; t < 4; ++t)
      o[((size_t)(b * SS + qg)) * DD + hd + t * 16 + n] = accO[t][r] * inv;
  }
}

// ---------------------------------------------------------------------------
// Routing: scores = h @ r (D x 16), top-4 + softmax.
// ---------------------------------------------------------------------------
__global__ __launch_bounds__(32) void k_route(const float* __restrict__ h,
                                              const float* __restrict__ r1,
                                              const float* __restrict__ r2,
                                              int* __restrict__ ti1, float* __restrict__ c1,
                                              int* __restrict__ ti2, float* __restrict__ c2) {
  __shared__ float sc[32];
  const int tok = blockIdx.x, t = threadIdx.x;
  const float* row = h + (size_t)tok * DD;
  const float* rm = (t < 16) ? r1 : r2;
  const int p = t & 15;
  float acc = 0.f;
  for (int d = 0; d < DD; ++d) acc += row[d] * rm[(size_t)d * PP + p];
  sc[t] = acc;
  __syncthreads();
  if (t == 0 || t == 16) {
    const float* s = &sc[t];
    int* ti = (t == 0) ? ti1 : ti2;
    float* cc = (t == 0) ? c1 : c2;
    int idx[KK]; float val[KK]; bool used[PP] = {};
    for (int k = 0; k < KK; ++k) {
      int bi = 0; float bv = -__builtin_inff();
      for (int pp = 0; pp < PP; ++pp)
        if (!used[pp] && s[pp] > bv) { bv = s[pp]; bi = pp; }
      used[bi] = true; idx[k] = bi; val[k] = bv;
    }
    float m = val[0], sum = 0.f, e4[KK];
    for (int k = 0; k < KK; ++k) { e4[k] = __expf(val[k] - m); sum += e4[k]; }
    for (int k = 0; k < KK; ++k) {
      ti[tok * KK + k] = idx[k];
      cc[tok * KK + k] = e4[k] / sum;
    }
  }
}

// ---------------------------------------------------------------------------
// Deterministic counting sort of (token,slot) entries by expert.
// ---------------------------------------------------------------------------
__global__ __launch_bounds__(32) void k_build(const int* __restrict__ ti1,
                                              const int* __restrict__ ti2,
                                              int* __restrict__ ent1, int* __restrict__ eof1, int* __restrict__ off1,
                                              int* __restrict__ ent2, int* __restrict__ eof2, int* __restrict__ off2) {
  __shared__ int cnt1[PP], cnt2[PP];
  const int t = threadIdx.x;
  if (t < PP) {
    int c = 0;
    for (int i = 0; i < NENT; ++i) if (ti1[i] == t) ++c;
    cnt1[t] = c;
  } else {
    int e = t - PP, c = 0;
    for (int i = 0; i < NENT; ++i) if (ti2[i] == e) ++c;
    cnt2[e] = c;
  }
  __syncthreads();
  if (t == 0) { int a = 0; for (int e = 0; e < PP; ++e) { off1[e] = a; a += cnt1[e]; } off1[PP] = a; }
  if (t == 1) { int a = 0; for (int e = 0; e < PP; ++e) { off2[e] = a; a += cnt2[e]; } off2[PP] = a; }
  __syncthreads();
  if (t < PP) {
    int e = t, pos = off1[e];
    for (int i = 0; i < NENT; ++i)
      if (ti1[i] == e) { ent1[pos] = i; eof1[i] = pos; ++pos; }
  } else {
    int e = t - PP, pos = off2[e];
    for (int i = 0; i < NENT; ++i)
      if (ti2[i] == e) { ent2[pos] = i; eof2[i] = pos; ++pos; }
  }
}

// ---------------------------------------------------------------------------
// Expert GEMM over routed entries: 64 entries x 128 cols per block.
// B tiles double-buffered through LDS via GLOBAL_LOAD_ASYNC_TO_LDS
// (ASYNCcnt, in-order completion) overlapping HBM fetch with cvt + WMMA.
// ---------------------------------------------------------------------------
__global__ __launch_bounds__(256) void k_fcexp(const float* __restrict__ act,
                                               const float* __restrict__ bank,
                                               const int* __restrict__ entries,
                                               const int* __restrict__ off,
                                               const float* __restrict__ coef,
                                               float* __restrict__ zbuf,
                                               int Kd, int Nd) {
  const int e = blockIdx.z, tile = blockIdx.y, fb = blockIdx.x * 128;
  const int beg = off[e], end = off[e + 1];
  const int ebase = beg + tile * 64;
  if (ebase >= end) return;
  __shared__ bf16  sA[64][32];
  __shared__ bf16  sBf[8][32][LSTR];
  __shared__ float sF[2][32][128];     // double-buffered async landing zone
  __shared__ int   rowTok[64];
  __shared__ float rowCf[64];
  const int tid = threadIdx.x;
  if (tid < 64) {
    int idx = ebase + tid;
    if (idx < end) {
      int ent = entries[idx];
      rowTok[tid] = ent >> 2;
      rowCf[tid] = coef[ent];
    } else { rowTok[tid] = -1; rowCf[tid] = 0.f; }
  }
  __syncthreads();
  const int wave = tid >> 5, lane = tid & 31;
  const int n = lane & 15, rbase = (lane < 16) ? 0 : 8;
  const int brow = tid >> 3, bcc = (tid & 7) * 16;  // async: 64B per thread
  const float* Bsrc = bank + (size_t)e * Kd * Nd + fb;
  v8f acc[4] = {};

  // prologue: async-fetch first B tile into buffer 0
  {
    const float* p = Bsrc + (size_t)brow * Nd + bcc;
    uint32_t dst = (uint32_t)(size_t)&sF[0][brow][bcc];
#pragma unroll
    for (int j = 0; j < 4; ++j) {
      uint64_t ga = (uint64_t)(size_t)(p + j * 4);
      uint32_t la = dst + j * 16;
      asm volatile("global_load_async_to_lds_b128 %0, %1, off"
                   :: "v"(la), "v"(ga) : "memory");
    }
  }
  int buf = 0;
  for (int k0 = 0; k0 < Kd; k0 += 32) {
    __syncthreads();  // prior WMMA readers of sA/sBf are done
    {  // A: coefficient-scaled gather, 8 consecutive elems -> 16B ds_store
      int r = tid >> 2, kc = (tid & 3) * 8;
      int tok = rowTok[r]; float cf = rowCf[r];
      v8bf t;
      if (tok >= 0) {
        const float* p = act + (size_t)tok * Kd + k0 + kc;
#pragma unroll
        for (int j = 0; j < 8; ++j) t[j] = f2bf(p[j] * cf);
      } else {
#pragma unroll
        for (int j = 0; j < 8; ++j) t[j] = f2bf(0.f);
      }
      st16B(&sA[r][kc], t);
    }
    // issue next tile's asyncs, then drain current tile (in-order per ISA)
    if (k0 + 32 < Kd) {
      const float* p = Bsrc + (size_t)(k0 + 32 + brow) * Nd + bcc;
      uint32_t dst = (uint32_t)(size_t)&sF[buf ^ 1][brow][bcc];
#pragma unroll
      for (int j = 0; j < 4; ++j) {
        uint64_t ga = (uint64_t)(size_t)(p + j * 4);
        uint32_t la = dst + j * 16;
        asm volatile("global_load_async_to_lds_b128 %0, %1, off"
                     :: "v"(la), "v"(ga) : "memory");
      }
      asm volatile("s_wait_asynccnt 0x4" ::: "memory");
    } else {
      asm volatile("s_wait_asynccnt 0x0" ::: "memory");
    }
    __syncthreads();
    {  // convert current buffer -> fragment-major bf16: 4k x 4n per thread
      int kg = (tid >> 5) * 4;
      int ng = (tid & 31) * 4;
      float ld[4][4];
#pragma unroll
      for (int i = 0; i < 4; ++i) {
        float4 v = *(const float4*)__builtin_assume_aligned(&sF[buf][kg + i][ng], 16);
        ld[i][0] = v.x; ld[i][1] = v.y; ld[i][2] = v.z; ld[i][3] = v.w;
      }
#pragma unroll
      for (int j = 0; j < 4; ++j) {
        int nn = ng + j;
        v4bf t;
#pragma unroll
        for (int i = 0; i < 4; ++i) t[i] = f2bf(ld[i][j]);
        st8B(&sBf[nn >> 4][BF_LANE(kg, nn)][kg & 15], t);
      }
    }
    __syncthreads();
    v16bf b = load_bf_frag(&sBf[wave][0][0]);
#pragma unroll
    for (int rf = 0; rf < 4; ++rf) {
      v16bf a = load_a_frag(&sA[rf * 16][0], 32);
      acc[rf] = wmma_bf16(a, b, acc[rf]);
    }
    buf ^= 1;
  }
#pragma unroll
  for (int rf = 0; rf < 4; ++rf) {
#pragma unroll
    for (int r = 0; r < 8; ++r) {
      int idx = ebase + rf * 16 + rbase + r;
      if (idx < end)
        zbuf[(size_t)idx * Nd + fb + wave * 16 + n] = acc[rf][r];
    }
  }
}

// ---------------------------------------------------------------------------
// Combine kernels (deterministic slot sums)
// ---------------------------------------------------------------------------
__global__ __launch_bounds__(256) void k_combine1(const float* __restrict__ zbuf,
                                                  const int* __restrict__ eof,
                                                  float* __restrict__ hmid) {
  const int tok = blockIdx.y;
  const int f = blockIdx.x * 256 + threadIdx.x;
  float s = 0.f;
#pragma unroll
  for (int k = 0; k < KK; ++k)
    s += zbuf[(size_t)eof[tok * KK + k] * FF + f];
  float u = 0.7978845608028654f * (s + 0.044715f * s * s * s);
  hmid[(size_t)tok * FF + f] = 0.5f * s * (1.0f + tanhf(u));
}

__global__ __launch_bounds__(256) void k_combine2(const float* __restrict__ zbuf,
                                                  const int* __restrict__ eof,
                                                  const float* __restrict__ x2,
                                                  float* __restrict__ out) {
  const int tok = blockIdx.y;
  const int d = blockIdx.x * 256 + threadIdx.x;
  float s = x2[(size_t)tok * DD + d];
#pragma unroll
  for (int k = 0; k < KK; ++k)
    s += zbuf[(size_t)eof[tok * KK + k] * DD + d];
  out[(size_t)tok * DD + d] = s;
}

// ---------------------------------------------------------------------------
// Launch
// ---------------------------------------------------------------------------
extern "C" void kernel_launch(void* const* d_in, const int* in_sizes, int n_in,
                              void* d_out, int out_size, void* d_ws, size_t ws_size,
                              hipStream_t stream) {
  const float* x    = (const float*)d_in[0];
  const float* wq   = (const float*)d_in[1];
  const float* wk   = (const float*)d_in[2];
  const float* wv   = (const float*)d_in[3];
  const float* wo   = (const float*)d_in[4];
  const float* g1   = (const float*)d_in[5];
  const float* g2   = (const float*)d_in[6];
  const float* r1   = (const float*)d_in[7];
  const float* r2   = (const float*)d_in[8];
  const float* bank1 = (const float*)d_in[9];
  const float* bank2 = (const float*)d_in[10];
  float* out = (float*)d_out;

  float* ws = (float*)d_ws;
  const size_t M1 = (size_t)NTOK * DD;
  float* h1    = ws;            ws += M1;
  float* qb    = ws;            ws += M1;
  float* kb    = ws;            ws += M1;
  float* vb    = ws;            ws += M1;
  float* ob    = ws;            ws += M1;
  float* x2    = ws;            ws += M1;
  float* h2    = ws;            ws += M1;
  float* hmid  = ws;            ws += (size_t)NTOK * FF;
  float* zbuf1 = ws;            ws += (size_t)NENT * FF;
  float* zbuf2 = ws;            ws += (size_t)NENT * DD;
  float* c1    = ws;            ws += NENT;
  float* c2    = ws;            ws += NENT;
  int* ip = (int*)ws;
  int* ti1  = ip;  ip += NENT;
  int* ti2  = ip;  ip += NENT;
  int* ent1 = ip;  ip += NENT;
  int* ent2 = ip;  ip += NENT;
  int* eof1 = ip;  ip += NENT;
  int* eof2 = ip;  ip += NENT;
  int* off1 = ip;  ip += PP + 1;
  int* off2 = ip;  ip += PP + 1;

  k_rmsnorm<<<NTOK, 256, 0, stream>>>(x, g1, h1);
  k_gemm<<<dim3(DD / 128, NTOK / 128), 256, 0, stream>>>(h1, wq, nullptr, qb, NTOK, DD, DD);
  k_gemm<<<dim3(DD / 128, NTOK / 128), 256, 0, stream>>>(h1, wk, nullptr, kb, NTOK, DD, DD);
  k_gemm<<<dim3(DD / 128, NTOK / 128), 256, 0, stream>>>(h1, wv, nullptr, vb, NTOK, DD, DD);
  k_attn<<<dim3(SS / 64, HH, BB), 128, 0, stream>>>(qb, kb, vb, ob);
  k_gemm<<<dim3(DD / 128, NTOK / 128), 256, 0, stream>>>(ob, wo, x, x2, NTOK, DD, DD);
  k_rmsnorm<<<NTOK, 256, 0, stream>>>(x2, g2, h2);
  k_route<<<NTOK, 32, 0, stream>>>(h2, r1, r2, ti1, c1, ti2, c2);
  k_build<<<1, 32, 0, stream>>>(ti1, ti2, ent1, eof1, off1, ent2, eof2, off2);
  k_fcexp<<<dim3(FF / 128, NTOK / 64, PP), 256, 0, stream>>>(
      h2, bank1, ent1, off1, c1, zbuf1, DD, FF);
  k_combine1<<<dim3(FF / 256, NTOK), 256, 0, stream>>>(zbuf1, eof1, hmid);
  k_fcexp<<<dim3(DD / 128, NTOK / 64, PP), 256, 0, stream>>>(
      hmid, bank2, ent2, off2, c2, zbuf2, FF, DD);
  k_combine2<<<dim3(DD / 256, NTOK), 256, 0, stream>>>(zbuf2, eof2, x2, out);
}
